// ShapedAttention_1168231104703
// MI455X (gfx1250) — compile-verified, matrix operand order checked by reading, not run on hardware
//
#include <hip/hip_runtime.h>
#include <hip/hip_bf16.h>

// ---------------------------------------------------------------------------
// Shaped attention, CDNA5 (gfx1250), wave32, WMMA f32_16x16x32_f16,
// async global->LDS staging (ASYNCcnt) for K/V tiles in the attention kernel.
// out = beta*(softmax(QK^T*s)@V) + alpha*V[q] - gamma*C*colsum(V)
// ---------------------------------------------------------------------------

typedef _Float16 half_t;
typedef __attribute__((ext_vector_type(16))) _Float16 v16h;
typedef __attribute__((ext_vector_type(8)))  float    v8f;
typedef __attribute__((ext_vector_type(4)))  unsigned int v4u;
typedef int v4i_ __attribute__((vector_size(16)));   // pointee type of async builtin

union Frag { v16h h; v4u u[2]; };

static __device__ inline v8f zero8() {
    v8f z = {0.f,0.f,0.f,0.f,0.f,0.f,0.f,0.f};
    return z;
}

#define WMMA16(A, B, C) \
    __builtin_amdgcn_wmma_f32_16x16x32_f16(false, (A), false, (B), (short)0, (C), false, false)

#ifndef __has_builtin
#define __has_builtin(x) 0
#endif

#if __has_builtin(__builtin_amdgcn_global_load_async_to_lds_b128)
#define HAVE_ASYNC 1
#else
#define HAVE_ASYNC 0
#endif

#if __has_builtin(__builtin_amdgcn_s_wait_asynccnt)
#define WAIT_ASYNC() __builtin_amdgcn_s_wait_asynccnt(0)
#elif HAVE_ASYNC
#define WAIT_ASYNC() asm volatile("s_wait_asynccnt 0" ::: "memory")
#else
#define WAIT_ASYNC() ((void)0)
#endif

// One 16B chunk per lane: global -> LDS (async if available).
// Builtin signature (from clang diagnostic): (as1 v4i*, as3 v4i*, Imm off, Imm cpol)
static __device__ __forceinline__ void async_b128(const half_t* gp, half_t* lp) {
#if HAVE_ASYNC
    __builtin_amdgcn_global_load_async_to_lds_b128(
        (__attribute__((address_space(1))) v4i_*)gp,
        (__attribute__((address_space(3))) v4i_*)lp, 0, 0);
#else
    *(v4u*)lp = *(const v4u*)gp;
#endif
}

#define NB   2
#define NT   2048
#define ND   768
#define NH   12
#define NHD  64
#define NBT  4096

// ---------------------------------------------------------------------------
// f32 -> f16 convert (grid-stride)
// ---------------------------------------------------------------------------
__global__ void cvt_f16(const float* __restrict__ src, half_t* __restrict__ dst, int n) {
    int i = blockIdx.x * blockDim.x + threadIdx.x;
    int stride = gridDim.x * blockDim.x;
    for (; i < n; i += stride) dst[i] = (half_t)src[i];
}

// ---------------------------------------------------------------------------
// QKV projection GEMM: y = x @ W^T + b  (W row-major [768,768])
// Block: 256 thr = 8 waves (2x4), wave tile 32x32, block tile 64x128.
// N dimension spans [Wq | Wk | Wv] = 2304 cols; each 128-col block hits one W.
// Q stored *0.125 (softmax scale folded), K row-major, V stored TRANSPOSED.
// ---------------------------------------------------------------------------
__global__ __launch_bounds__(256) void qkv_gemm(
    const half_t* __restrict__ xh,    // [4096,768]
    const half_t* __restrict__ wh,    // [3][768][768]
    const float* __restrict__ bq, const float* __restrict__ bk,
    const float* __restrict__ bv,
    half_t* __restrict__ Qh,          // [B,H,T,64]
    half_t* __restrict__ Kh,          // [B,H,T,64]
    half_t* __restrict__ Vt)          // [B,H,64,T]
{
    const int lane = threadIdx.x & 31;
    const int wid  = threadIdx.x >> 5;
    const int ln16 = lane & 15;
    const int g    = lane >> 4;

    const int bm = blockIdx.x / 18;
    const int bn = blockIdx.x % 18;
    const int wm = wid >> 2;
    const int wn = wid & 3;
    const int Mbase = bm * 64 + wm * 32;
    const int Nbase = bn * 128 + wn * 32;
    const int which = Nbase / ND;       // 0=q 1=k 2=v
    const int nw0   = Nbase % ND;

    const half_t* wsrc = wh + (size_t)which * ND * ND;
    const float*  bias = (which == 0) ? bq : ((which == 1) ? bk : bv);

    v8f acc[2][2];
    acc[0][0] = zero8(); acc[0][1] = zero8();
    acc[1][0] = zero8(); acc[1][1] = zero8();

    const half_t* arow0 = xh + (size_t)(Mbase + ln16) * ND;
    const half_t* arow1 = arow0 + (size_t)16 * ND;
    const half_t* brow0 = wsrc + (size_t)(nw0 + ln16) * ND;
    const half_t* brow1 = brow0 + (size_t)16 * ND;

    for (int kc = 0; kc < ND; kc += 32) {
        Frag a0, a1, b0, b1;
        a0.u[0] = *(const v4u*)(arow0 + kc + 8 * g);
        a0.u[1] = *(const v4u*)(arow0 + kc + 16 + 8 * g);
        a1.u[0] = *(const v4u*)(arow1 + kc + 8 * g);
        a1.u[1] = *(const v4u*)(arow1 + kc + 16 + 8 * g);
        b0.u[0] = *(const v4u*)(brow0 + kc + 8 * g);
        b0.u[1] = *(const v4u*)(brow0 + kc + 16 + 8 * g);
        b1.u[0] = *(const v4u*)(brow1 + kc + 8 * g);
        b1.u[1] = *(const v4u*)(brow1 + kc + 16 + 8 * g);
        acc[0][0] = WMMA16(a0.h, b0.h, acc[0][0]);
        acc[0][1] = WMMA16(a0.h, b1.h, acc[0][1]);
        acc[1][0] = WMMA16(a1.h, b0.h, acc[1][0]);
        acc[1][1] = WMMA16(a1.h, b1.h, acc[1][1]);
    }

    // Epilogue. C/D layout: lane holds col n = ln16, rows r+8g (r = vgpr idx).
    for (int mt = 0; mt < 2; ++mt) {
        const int tok0 = Mbase + mt * 16 + 8 * g;   // 8 consecutive tokens
        const int b    = tok0 / NT;
        const int t0   = tok0 % NT;
        for (int nt = 0; nt < 2; ++nt) {
            const int nwc = nw0 + nt * 16 + ln16;   // col within this W
            const int h = nwc >> 6, d = nwc & 63;
            const float bb = bias[nwc];
            if (which == 2) {
                // V: transpose store -> Vt[(b*H+h)*64+d][t0..t0+7], one b128
                union { half_t hv[8]; v4u u; } pk;
                #pragma unroll
                for (int r = 0; r < 8; ++r) pk.hv[r] = (half_t)(acc[mt][nt][r] + bb);
                *(v4u*)(Vt + ((size_t)((b * NH + h) * NHD + d)) * NT + t0) = pk.u;
            } else {
                half_t* dst = (which == 0) ? Qh : Kh;
                const float s = (which == 0) ? 0.125f : 1.0f;  // hd^-0.5 folded into Q
                const size_t base = ((size_t)(b * NH + h) * NT) * NHD + d;
                #pragma unroll
                for (int r = 0; r < 8; ++r)
                    dst[base + (size_t)(t0 + r) * NHD] = (half_t)((acc[mt][nt][r] + bb) * s);
            }
        }
    }
}

// ---------------------------------------------------------------------------
// colsum[b,h,d] = sum_t V[b,h,t,d]   (reads transposed Vt rows, contiguous)
// ---------------------------------------------------------------------------
__global__ __launch_bounds__(256) void v_colsum(
    const half_t* __restrict__ Vt, float* __restrict__ cs)
{
    const int bh  = blockIdx.x;            // 0..23
    const int d   = threadIdx.x & 63;
    const int seg = threadIdx.x >> 6;      // 4 segments over T
    const half_t* row = Vt + ((size_t)bh * NHD + d) * NT;
    float s = 0.f;
    for (int t = seg * 512; t < seg * 512 + 512; ++t) s += (float)row[t];
    __shared__ float red[4][64];
    red[seg][d] = s;
    __syncthreads();
    if (seg == 0) cs[(size_t)bh * NHD + d] = red[0][d] + red[1][d] + red[2][d] + red[3][d];
}

// ---------------------------------------------------------------------------
// Flash attention + shaped epilogue.
// Block = 8 waves, all on one (b,h); 128 queries per block (16 per wave).
// K tile (32x64) and V^T tile (64x32) staged in LDS via async loads,
// double-buffered, 16B chunk per lane, 2 async instrs per wave per step.
// ---------------------------------------------------------------------------
__global__ __launch_bounds__(256) void shaped_attn(
    const half_t* __restrict__ Qh, const half_t* __restrict__ Kh,
    const half_t* __restrict__ Vt, const float* __restrict__ colsum,
    const float* __restrict__ alpha, const float* __restrict__ beta,
    const float* __restrict__ gamma, const float* __restrict__ Cbuf,
    float* __restrict__ out)
{
    const int lane = threadIdx.x & 31;
    const int wid  = threadIdx.x >> 5;
    const int ln16 = lane & 15;
    const int g    = lane >> 4;

    const int bh = blockIdx.x >> 4;            // b*H + h   (24 values)
    const int qc = blockIdx.x & 15;            // 128-query chunk
    const int qb = qc * 128 + wid * 16;        // this wave's first query row
    const int h  = bh % NH;
    const int b  = bh / NH;

    const half_t* Qbase = Qh + ((size_t)bh * NT) * NHD;
    const half_t* Kbase = Kh + ((size_t)bh * NT) * NHD;
    const half_t* Vbase = Vt + ((size_t)bh * NHD) * NT;

    // padded LDS tiles (pitches 72/40 halves: 16B-aligned rows, no bank conflicts)
    __shared__ __attribute__((aligned(16))) half_t Kl[2][32 * 72];  // 32 keys x 64 d
    __shared__ __attribute__((aligned(16))) half_t Vl[2][64 * 40];  // 64 d x 32 keys
    __shared__ __attribute__((aligned(16))) half_t Pl[8][16 * 40];  // per-wave P
    half_t* P = Pl[wid];

    // cooperative staging: 256 chunks of 16B per tile, chunk = wid*32+lane
    const int ck   = wid * 32 + lane;
    const int krow = ck >> 3, kcol = (ck & 7) * 8;   // K tile rows of 64 halves
    const int vrow = ck >> 2, vcol = (ck & 3) * 8;   // V tile rows of 32 halves

    // Q fragments (rows qb+ln16, full hd=64 as two K=32 frags)
    const half_t* qrow = Qbase + (size_t)(qb + ln16) * NHD;
    Frag qf[2];
    qf[0].u[0] = *(const v4u*)(qrow + 8 * g);
    qf[0].u[1] = *(const v4u*)(qrow + 16 + 8 * g);
    qf[1].u[0] = *(const v4u*)(qrow + 32 + 8 * g);
    qf[1].u[1] = *(const v4u*)(qrow + 48 + 8 * g);

    float mrow[8], lrow[8];
    #pragma unroll
    for (int r = 0; r < 8; ++r) { mrow[r] = -1e30f; lrow[r] = 0.f; }
    v8f O[4];
    #pragma unroll
    for (int nt = 0; nt < 4; ++nt) O[nt] = zero8();

    // preload first K/V tiles into buffer 0
    async_b128(Kbase + (size_t)krow * NHD + kcol, &Kl[0][krow * 72 + kcol]);
    async_b128(Vbase + (size_t)vrow * NT + vcol,  &Vl[0][vrow * 40 + vcol]);

    int cur = 0;
    for (int kb = 0; kb < NT; kb += 32) {
        WAIT_ASYNC();          // this wave's staged chunks are in LDS
        __syncthreads();       // whole tile (all waves' chunks) visible

        if (kb + 32 < NT) {    // prefetch next tile into the other buffer
            const int nxt = cur ^ 1;
            async_b128(Kbase + (size_t)(kb + 32 + krow) * NHD + kcol,
                       &Kl[nxt][krow * 72 + kcol]);
            async_b128(Vbase + (size_t)vrow * NT + (kb + 32) + vcol,
                       &Vl[nxt][vrow * 40 + vcol]);
        }

        // ---- S = Q*K^T for 16x32 key block (two 16x16 n-tiles) ----
        v8f S0 = zero8(), S1 = zero8();
        const half_t* kr0 = &Kl[cur][ln16 * 72];
        const half_t* kr1 = &Kl[cur][(16 + ln16) * 72];
        #pragma unroll
        for (int kc = 0; kc < 2; ++kc) {
            Frag b0, b1;
            b0.u[0] = *(const v4u*)(kr0 + kc * 32 + 8 * g);
            b0.u[1] = *(const v4u*)(kr0 + kc * 32 + 16 + 8 * g);
            b1.u[0] = *(const v4u*)(kr1 + kc * 32 + 8 * g);
            b1.u[1] = *(const v4u*)(kr1 + kc * 32 + 16 + 8 * g);
            S0 = WMMA16(qf[kc].h, b0.h, S0);
            S1 = WMMA16(qf[kc].h, b1.h, S1);
        }

        // ---- online softmax; row m = r+8g lives across the 16 lanes of half g
        float corr[8];
        #pragma unroll
        for (int r = 0; r < 8; ++r) {
            float bmax = fmaxf(S0[r], S1[r]);
            #pragma unroll
            for (int off = 1; off < 16; off <<= 1)
                bmax = fmaxf(bmax, __shfl_xor(bmax, off, 32));
            const float mnew = fmaxf(mrow[r], bmax);
            const float e0 = __expf(S0[r] - mnew);
            const float e1 = __expf(S1[r] - mnew);
            float bsum = e0 + e1;
            #pragma unroll
            for (int off = 1; off < 16; off <<= 1)
                bsum += __shfl_xor(bsum, off, 32);
            const float c = __expf(mrow[r] - mnew);
            lrow[r] = lrow[r] * c + bsum;
            mrow[r] = mnew;
            corr[r] = c;
            // stage P (C-layout -> row-major LDS) for the A-frag reload
            P[(r + 8 * g) * 40 + ln16]      = (half_t)e0;
            P[(r + 8 * g) * 40 + 16 + ln16] = (half_t)e1;
        }
        #pragma unroll
        for (int nt = 0; nt < 4; ++nt)
            #pragma unroll
            for (int r = 0; r < 8; ++r) O[nt][r] *= corr[r];

        // per-wave P tile: same-wave DS ops are in-order; force completion +
        // compiler ordering before the transposed re-read.
        asm volatile("s_wait_dscnt 0" ::: "memory");

        // ---- O += P(16x32) * V(32x64) : 4 n-tiles ----
        Frag pf;
        const half_t* prow = P + ln16 * 40;
        pf.u[0] = *(const v4u*)(prow + 8 * g);
        pf.u[1] = *(const v4u*)(prow + 16 + 8 * g);
        #pragma unroll
        for (int nt = 0; nt < 4; ++nt) {
            const half_t* vr = &Vl[cur][(nt * 16 + ln16) * 40];
            Frag vf;
            vf.u[0] = *(const v4u*)(vr + 8 * g);
            vf.u[1] = *(const v4u*)(vr + 16 + 8 * g);
            O[nt] = WMMA16(pf.h, vf.h, O[nt]);
        }

        __syncthreads();       // everyone done reading buf[cur] before reuse
        cur ^= 1;
    }

    // ---- shaped epilogue ----
    const float al = alpha[h], be = beta[h], ga = gamma[h], Cv = Cbuf[h];
    const float* csrow = colsum + (size_t)bh * NHD;
    #pragma unroll
    for (int nt = 0; nt < 4; ++nt) {
        const int d = nt * 16 + ln16;
        const float gterm = ga * Cv * csrow[d];
        union { v4u u; half_t hv[8]; } vq;   // V[qb+8g .. +7][d] via Vt row (contig)
        vq.u = *(const v4u*)(Vbase + (size_t)d * NT + qb + 8 * g);
        #pragma unroll
        for (int r = 0; r < 8; ++r) {
            const int t = qb + 8 * g + r;
            const float o = be * (O[nt][r] / lrow[r]) + al * (float)vq.hv[r] - gterm;
            out[((size_t)b * NT + t) * ND + h * NHD + d] = o;
        }
    }
}

// ---------------------------------------------------------------------------
extern "C" void kernel_launch(void* const* d_in, const int* in_sizes, int n_in,
                              void* d_out, int out_size, void* d_ws, size_t ws_size,
                              hipStream_t stream) {
    const float* x     = (const float*)d_in[0];
    const float* Wq    = (const float*)d_in[1];
    const float* bq    = (const float*)d_in[2];
    const float* Wk    = (const float*)d_in[3];
    const float* bk    = (const float*)d_in[4];
    const float* Wv    = (const float*)d_in[5];
    const float* bv    = (const float*)d_in[6];
    const float* alpha = (const float*)d_in[7];
    const float* beta  = (const float*)d_in[8];
    const float* gamma = (const float*)d_in[9];
    const float* Cbuf  = (const float*)d_in[10];
    float* out = (float*)d_out;

    // workspace layout (bytes)
    char* ws = (char*)d_ws;
    half_t* xh = (half_t*)(ws);                       //  6,291,456
    half_t* Wh = (half_t*)(ws + 6291456);             //  3,538,944 (Wq|Wk|Wv f16)
    half_t* Qh = (half_t*)(ws + 9830400);             //  6,291,456
    half_t* Kh = (half_t*)(ws + 16121856);            //  6,291,456
    half_t* Vt = (half_t*)(ws + 22413312);            //  6,291,456 (transposed)
    float*  cs = (float*) (ws + 28704768);            //      6,144

    const int NW = ND * ND;  // 589824

    cvt_f16<<<2048, 256, 0, stream>>>(x,  xh, NBT * ND);
    cvt_f16<<<1024, 256, 0, stream>>>(Wq, Wh,          NW);
    cvt_f16<<<1024, 256, 0, stream>>>(Wk, Wh + NW,     NW);
    cvt_f16<<<1024, 256, 0, stream>>>(Wv, Wh + 2 * NW, NW);

    qkv_gemm<<<1152, 256, 0, stream>>>(xh, Wh, bq, bk, bv, Qh, Kh, Vt);
    v_colsum<<<NB * NH, 256, 0, stream>>>(Vt, cs);
    shaped_attn<<<384, 256, 0, stream>>>(Qh, Kh, Vt, cs, alpha, beta, gamma, Cbuf, out);

    (void)in_sizes; (void)n_in; (void)out_size; (void)ws_size;
}
